// CNF2_52931176956415
// MI455X (gfx1250) — matrix operand order repaired and checked by reading, not compiled, not run on hardware
//
#include <hip/hip_runtime.h>
#include <stdint.h>

#define IN_OUT_DIM 64
#define HIDDEN_DIM 512
#define WIDTH 128
#define BATCH 262144
#define BLOCKSIZE (WIDTH * IN_OUT_DIM)
#define P_TOTAL (2 * BLOCKSIZE + WIDTH) /* 16512 */
#define TILES_PER_WAVE 4                /* 4 waves * 4 tiles * 16 rows = 256/block */

typedef __attribute__((ext_vector_type(16))) _Float16 v16h;
typedef __attribute__((ext_vector_type(8)))  float    v8f;

union Frag {
  v16h     h;
  uint32_t u[8];
  uint4    q[2];
};

/* workspace layout (dword offsets) */
#define WS_HIDDEN 0
#define WS_W      512
#define WS_U      (512 + 8192)
#define WS_BIAS   (512 + 8192 + 8192)
#define WS_S      (WS_BIAS + 128)
#define WS_AFRAG  (WS_S + 128)        /* 4096 dwords: W as WMMA A-fragments */
#define WS_BFRAG  (WS_AFRAG + 4096)   /* 4096 dwords: U as WMMA B-fragments */

__device__ __forceinline__ uint32_t pk_f16(float a, float b) {
  union { _Float16 h[2]; uint32_t u; } x;
  x.h[0] = (_Float16)a; x.h[1] = (_Float16)b;
  return x.u;
}

/* Branch-free tanh: native v_tanh_f32 on gfx1250, else exp2/rcp identity. */
__device__ __forceinline__ float fast_tanh(float x) {
#if __has_builtin(__builtin_amdgcn_tanhf)
  return __builtin_amdgcn_tanhf(x);
#else
  /* tanh(x) = 1 - 2/(exp(2x)+1); exp(2x) = exp2(x * 2*log2(e)) */
  float e = __builtin_amdgcn_exp2f(x * 2.885390082f);
  return 1.f - 2.f * __builtin_amdgcn_rcpf(e + 1.f);
#endif
}

/* ---- hypernet stage 1: hidden = tanh(t * fc1_w + fc1_b), 512 values ---- */
__global__ void cnf_hyper_hidden(const float* __restrict__ t,
                                 const float* __restrict__ fc1_w,
                                 const float* __restrict__ fc1_b,
                                 float* __restrict__ ws) {
  int i = threadIdx.x;
  ws[WS_HIDDEN + i] = fast_tanh(t[0] * fc1_w[i] + fc1_b[i]);
}

/* ---- hypernet stage 2: p = hidden @ fc3_w^T + fc3_b -> W, U, Bias ---- */
__global__ void cnf_hyper_fc3(const float* __restrict__ fc3_w,
                              const float* __restrict__ fc3_b,
                              float* __restrict__ ws) {
  __shared__ float sh[HIDDEN_DIM];
  for (int k = threadIdx.x; k < HIDDEN_DIM; k += blockDim.x)
    sh[k] = ws[WS_HIDDEN + k];
  __syncthreads();
  int o = blockIdx.x * blockDim.x + threadIdx.x;
  if (o >= P_TOTAL) return;
  const float4* row = (const float4*)(fc3_w + (size_t)o * HIDDEN_DIM);
  float acc = fc3_b[o];
#pragma unroll 4
  for (int k4 = 0; k4 < HIDDEN_DIM / 4; ++k4) {
    float4 wv = row[k4];
    acc += wv.x * sh[k4 * 4 + 0] + wv.y * sh[k4 * 4 + 1] +
           wv.z * sh[k4 * 4 + 2] + wv.w * sh[k4 * 4 + 3];
  }
  if (o < BLOCKSIZE)            ws[WS_W + o] = acc;
  else if (o < 2 * BLOCKSIZE)   ws[WS_U + (o - BLOCKSIZE)] = acc;
  else                          ws[WS_BIAS + (o - 2 * BLOCKSIZE)] = acc;
}

/* ---- pre-swizzle W/U into WMMA fragment layouts, compute s = sum W*U ---- */
__global__ void cnf_prep_frags(float* __restrict__ wsf) {
  uint32_t* ws = (uint32_t*)wsf;
  const float* W = wsf + WS_W;
  const float* U = wsf + WS_U;
  int tid = threadIdx.x;
  if (tid < WIDTH) {
    float s = 0.f;
#pragma unroll
    for (int d = 0; d < IN_OUT_DIM; ++d)
      s += W[tid * IN_OUT_DIM + d] * U[tid * IN_OUT_DIM + d];
    wsf[WS_S + tid] = s;
  }
  /* A fragments (16-bit A 16x32 layout): idx = ((m*2+kt)*32 + lane)*8 + v */
  for (int idx = tid; idx < 4096; idx += blockDim.x) {
    int v = idx & 7, l = (idx >> 3) & 31, kt = (idx >> 8) & 1, m = idx >> 9;
    int M  = m * 16 + (l & 15);
    int k0 = kt * 32 + ((v < 4) ? 0 : 16) + ((l < 16) ? 0 : 8) + 2 * (v & 3);
    ws[WS_AFRAG + idx] = pk_f16(W[M * IN_OUT_DIM + k0], W[M * IN_OUT_DIM + k0 + 1]);
  }
  /* B fragments (16-bit B 32x16): idx = ((kt2*4+nt)*32 + l)*8 + v */
  for (int idx = tid; idx < 4096; idx += blockDim.x) {
    int v = idx & 7, l = (idx >> 3) & 31, nt = (idx >> 8) & 3, kt2 = idx >> 10;
    int N  = nt * 16 + (l & 15);
    int k0 = kt2 * 32 + ((v < 4) ? 0 : 16) + ((l < 16) ? 0 : 8) + 2 * (v & 3);
    ws[WS_BFRAG + idx] = pk_f16(U[k0 * IN_OUT_DIM + N], U[(k0 + 1) * IN_OUT_DIM + N]);
  }
}

/* ---- main fused kernel: h=tanh(W z^T + b); dz = h^T U /128; trace ---- */
__global__ void __launch_bounds__(128)
cnf_main(const float* __restrict__ z,
         const float* __restrict__ wsf,
         float* __restrict__ out_dz,
         float* __restrict__ out_logp) {
  __shared__ __align__(16) uint32_t sA[4096];
  __shared__ __align__(16) uint32_t sB[4096];
  __shared__ float sBias[WIDTH];
  __shared__ float sS[WIDTH];
  const uint32_t* ws = (const uint32_t*)wsf;
  int tid = threadIdx.x;
  { /* stage fragments + bias/s into LDS once per block (amortized over 4 tiles/wave) */
    const uint4* ga = (const uint4*)(ws + WS_AFRAG);
    const uint4* gb = (const uint4*)(ws + WS_BFRAG);
    uint4* la = (uint4*)sA;
    uint4* lb = (uint4*)sB;
#pragma unroll
    for (int i = 0; i < 8; ++i) {
      la[tid + i * 128] = ga[tid + i * 128];
      lb[tid + i * 128] = gb[tid + i * 128];
    }
    if (tid < WIDTH) { sBias[tid] = wsf[WS_BIAS + tid]; sS[tid] = wsf[WS_S + tid]; }
  }
  __syncthreads();

  const int lane   = tid & 31;
  const int wave   = tid >> 5;
  const int col    = lane & 15;
  const int hiHalf = (lane >= 16) ? 1 : 0;
  const int wbias  = hiHalf ? 8 : 0;
  const float scale = 1.f / WIDTH;

  for (int tt = 0; tt < TILES_PER_WAVE; ++tt) {
    const int b0 = blockIdx.x * (128 * TILES_PER_WAVE / 2) +
                   wave * (16 * TILES_PER_WAVE) + tt * 16;

    /* z fragments: B operand of GEMM1 (K=d, N=batch). 64B coalesced per lane. */
    Frag zf[2];
#pragma unroll
    for (int dk = 0; dk < 2; ++dk) {
      const float4* zr = (const float4*)(z + (size_t)(b0 + col) * IN_OUT_DIM +
                                         dk * 32 + (hiHalf ? 8 : 0));
#pragma unroll
      for (int q4 = 0; q4 < 4; ++q4) {
        float4 zv = zr[q4 < 2 ? q4 : q4 + 2];
        zf[dk].h[q4 * 4 + 0] = (_Float16)zv.x;
        zf[dk].h[q4 * 4 + 1] = (_Float16)zv.y;
        zf[dk].h[q4 * 4 + 2] = (_Float16)zv.z;
        zf[dk].h[q4 * 4 + 3] = (_Float16)zv.w;
      }
    }

    v8f D0 = {}, D1 = {}, D2 = {}, D3 = {};
    float tracc = 0.f;

#pragma unroll
    for (int kt2 = 0; kt2 < 4; ++kt2) {      /* GEMM2 K-tile = w in [kt2*32, +32) */
      const int m0 = kt2 * 2, m1 = m0 + 1;   /* two GEMM1 w-row tiles */
      v8f C0 = {}, C1 = {};
#pragma unroll
      for (int dk = 0; dk < 2; ++dk) {       /* GEMM1 K over d = 64 */
        Frag a0, a1;
        const uint4* pa0 = (const uint4*)&sA[((m0 * 2 + dk) * 32 + lane) * 8];
        const uint4* pa1 = (const uint4*)&sA[((m1 * 2 + dk) * 32 + lane) * 8];
        a0.q[0] = pa0[0]; a0.q[1] = pa0[1];
        a1.q[0] = pa1[0]; a1.q[1] = pa1[1];
        C0 = __builtin_amdgcn_wmma_f32_16x16x32_f16(false, a0.h, false, zf[dk].h,
                                                    (short)0, C0, false, false);
        C1 = __builtin_amdgcn_wmma_f32_16x16x32_f16(false, a1.h, false, zf[dk].h,
                                                    (short)0, C1, false, false);
      }
      /* bias + tanh + trace; C layout == GEMM2 A layout -> per-lane repack only */
      Frag A2;
      float tpart = 0.f;
      const int wb = kt2 * 32 + wbias;
#pragma unroll
      for (int r = 0; r < 8; ++r) {
        const int w0 = wb + r, w1 = wb + 16 + r;
        float t0 = fast_tanh(C0[r] + sBias[w0]);
        float t1 = fast_tanh(C1[r] + sBias[w1]);
        tpart += (1.f - t0 * t0) * sS[w0] + (1.f - t1 * t1) * sS[w1];
        A2.h[r]     = (_Float16)t0;
        A2.h[r + 8] = (_Float16)t1;
      }
      tracc += tpart;
      /* GEMM2 partials over the 64 output dims (4 N tiles) */
      Frag bu;
      const uint4* pb;
      pb = (const uint4*)&sB[((kt2 * 4 + 0) * 32 + lane) * 8];
      bu.q[0] = pb[0]; bu.q[1] = pb[1];
      D0 = __builtin_amdgcn_wmma_f32_16x16x32_f16(false, A2.h, false, bu.h, (short)0, D0, false, false);
      pb = (const uint4*)&sB[((kt2 * 4 + 1) * 32 + lane) * 8];
      bu.q[0] = pb[0]; bu.q[1] = pb[1];
      D1 = __builtin_amdgcn_wmma_f32_16x16x32_f16(false, A2.h, false, bu.h, (short)0, D1, false, false);
      pb = (const uint4*)&sB[((kt2 * 4 + 2) * 32 + lane) * 8];
      bu.q[0] = pb[0]; bu.q[1] = pb[1];
      D2 = __builtin_amdgcn_wmma_f32_16x16x32_f16(false, A2.h, false, bu.h, (short)0, D2, false, false);
      pb = (const uint4*)&sB[((kt2 * 4 + 3) * 32 + lane) * 8];
      bu.q[0] = pb[0]; bu.q[1] = pb[1];
      D3 = __builtin_amdgcn_wmma_f32_16x16x32_f16(false, A2.h, false, bu.h, (short)0, D3, false, false);
    }

    /* trace: lane j covers w in {+0..7, +16..23}; lane 16+j the complement */
    float trace = (tracc + __shfl_xor(tracc, 16, 32)) * scale;
    if (lane < 16) out_logp[b0 + lane] = -trace;

#pragma unroll
    for (int r = 0; r < 8; ++r) {
      const int b = b0 + r + (hiHalf ? 8 : 0);
      float* orow = out_dz + (size_t)b * IN_OUT_DIM + col;
      orow[0]  = D0[r] * scale;
      orow[16] = D1[r] * scale;
      orow[32] = D2[r] * scale;
      orow[48] = D3[r] * scale;
    }
  }
}

extern "C" void kernel_launch(void* const* d_in, const int* in_sizes, int n_in,
                              void* d_out, int out_size, void* d_ws, size_t ws_size,
                              hipStream_t stream) {
  const float* t     = (const float*)d_in[0];
  const float* z     = (const float*)d_in[1];
  /* d_in[2] = logp_z, unused by the reference outputs */
  const float* fc1_w = (const float*)d_in[3];
  const float* fc1_b = (const float*)d_in[4];
  const float* fc3_w = (const float*)d_in[5];
  const float* fc3_b = (const float*)d_in[6];
  float* out = (float*)d_out;
  float* ws  = (float*)d_ws;

  cnf_hyper_hidden<<<1, HIDDEN_DIM, 0, stream>>>(t, fc1_w, fc1_b, ws);
  cnf_hyper_fc3<<<(P_TOTAL + 127) / 128, 128, 0, stream>>>(fc3_w, fc3_b, ws);
  cnf_prep_frags<<<1, 256, 0, stream>>>(ws);
  cnf_main<<<BATCH / (64 * TILES_PER_WAVE), 128, 0, stream>>>(
      z, ws, out, out + (size_t)BATCH * IN_OUT_DIM);
}
